// DeformableConv2d_70703751627014
// MI455X (gfx1250) — compile-verified, matrix operand order checked by reading, not run on hardware
//
#include <hip/hip_runtime.h>
#include <hip/hip_bf16.h>

typedef __attribute__((ext_vector_type(16))) __bf16 v16bf;
typedef __attribute__((ext_vector_type(8)))  float  v8f;
typedef __attribute__((ext_vector_type(4)))  unsigned int v4u;
typedef __attribute__((ext_vector_type(8)))  unsigned int v8u;

#define HWD   16384   // 128*128
#define WD    128
#define HD    128
#define CC    64
#define OC    64
#define NKC   18      // 576 / 32 K-chunks

static __device__ __forceinline__ unsigned short f2bf(float f) {
  unsigned u = __builtin_bit_cast(unsigned, f);
  u += 0x7FFFu + ((u >> 16) & 1u);          // round-to-nearest-even
  return (unsigned short)(u >> 16);
}
static __device__ __forceinline__ float bf2f(unsigned short h) {
  return __builtin_bit_cast(float, ((unsigned)h) << 16);
}

// ---------------------------------------------------------------------------
// Kernel 1: pre-swizzle weights into WMMA A-fragment layout, hi/lo bf16.
// Layout: wfrag[kk(9)][plane(2)][chunk(2)][mt(4)][lane(32)][16] ushort
// -> per-tap block is one contiguous 16 KB tile (TDM-friendly).
// Lane l (<16): element j -> k_local = (j<8? j : j+8); lane>=16: +8.
// K_global = kk*64 + c.
// ---------------------------------------------------------------------------
__global__ __launch_bounds__(256) void wprep_kernel(
    const float* __restrict__ weight,
    unsigned short* __restrict__ wfrag) {
  int g = blockIdx.x * 256 + threadIdx.x;
  if (g >= NKC * 4 * 32) return;
  int kc   = g >> 7;
  int rem  = g & 127;
  int mt   = rem >> 5;
  int lane = rem & 31;
  int kk   = kc >> 1;
  int chunk = kc & 1;
  int m    = mt * 16 + (lane & 15);
  int hb   = (lane >> 4) * 8;
  size_t basehi = ((((size_t)(kk * 2 + 0) * 2 + chunk) * 4 + mt) * 32 + lane) * 16;
  size_t baselo = ((((size_t)(kk * 2 + 1) * 2 + chunk) * 4 + mt) * 32 + lane) * 16;
#pragma unroll
  for (int j = 0; j < 16; ++j) {
    int kl = hb + (j < 8 ? j : j + 8);
    int Kg = kc * 32 + kl;
    int kkk = Kg >> 6;         // tap index 0..8
    int c   = Kg & 63;         // channel
    float wv = weight[(m * CC + c) * 9 + kkk];
    unsigned short h = f2bf(wv);
    unsigned short l = f2bf(wv - bf2f(h));
    wfrag[basehi + j] = h;
    wfrag[baselo + j] = l;
  }
}

// ---------------------------------------------------------------------------
// Kernel 2: offset-predictor conv (standard 3x3, pad=1) -> off (B,18,Ho,Wo)
// ---------------------------------------------------------------------------
__global__ __launch_bounds__(256) void offconv_kernel(
    const float* __restrict__ x, const float* __restrict__ ow,
    const float* __restrict__ ob, float* __restrict__ off) {
  int p  = blockIdx.x * 256 + threadIdx.x;    // 131072 pixels
  int b  = p >> 14;
  int hw = p & (HWD - 1);
  int ho = hw >> 7;
  int wo = hw & 127;
  float acc[18];
#pragma unroll
  for (int j = 0; j < 18; ++j) acc[j] = ob[j];
  const float* xb = x + (size_t)b * CC * HWD;
  for (int c = 0; c < CC; ++c) {
    float v[9];
#pragma unroll
    for (int kh = 0; kh < 3; ++kh)
#pragma unroll
      for (int kw = 0; kw < 3; ++kw) {
        int iy = ho - 1 + kh, ix = wo - 1 + kw;
        bool in = (iy >= 0) & (iy < HD) & (ix >= 0) & (ix < WD);
        v[kh * 3 + kw] = in ? xb[c * HWD + iy * WD + ix] : 0.f;
      }
    const float* wr = ow + c * 9;
#pragma unroll
    for (int j = 0; j < 18; ++j) {
      const float* wj = wr + j * (CC * 9);
#pragma unroll
      for (int t = 0; t < 9; ++t) acc[j] = fmaf(wj[t], v[t], acc[j]);
    }
  }
  float* op = off + (size_t)b * 18 * HWD + hw;
#pragma unroll
  for (int j = 0; j < 18; ++j) op[j * HWD] = acc[j];
}

// ---------------------------------------------------------------------------
// Kernel 3 (fused): TDM stages per-tap weight fragments into LDS (overlapped
// with bilinear gather), sampled hi/lo bf16 -> LDS B-fragments, then
// v_wmma_f32_16x16x32_bf16 3-term split GEMM -> bias -> store.
// One workgroup per (b, ho) row: N_tile = 128 pixels, M = 64 outputs.
// ---------------------------------------------------------------------------
__global__ __launch_bounds__(256) void dconv_wmma_kernel(
    const float* __restrict__ x, const float* __restrict__ bias,
    const float* __restrict__ off,
    const unsigned short* __restrict__ wfrag,
    float* __restrict__ out) {
  // B fragments: [plane hi/lo][c-chunk][lane-half][n][16] : 32 KB
  __shared__ unsigned short S[2][2][2][128][16];
  // A fragments (one tap): [plane][chunk][mt][lane][16] : 16 KB (TDM dest)
  __shared__ unsigned short A_lds[2][2][4][32][16];

  int t    = threadIdx.x;
  int b    = blockIdx.x >> 7;
  int ho   = blockIdx.x & 127;
  int wave = t >> 5;
  int lane = t & 31;
  int n_s  = t & 127;      // sampling role: pixel
  int ch_s = t >> 7;       // sampling role: channel half (0/1)

  v8f acc[4] = {};

  const float* offb = off + (size_t)b * 18 * HWD + ho * WD + n_s;
  unsigned lds_base = (unsigned)(size_t)(&A_lds[0][0][0][0][0]);

  for (int kk = 0; kk < 9; ++kk) {
    __syncthreads();   // previous S/A_lds consumers done

    // ---- wave 0: kick TDM copy of this tap's 16 KB fragment block ----
    if (wave == 0) {
      unsigned long long gaddr =
          (unsigned long long)(size_t)wfrag + (unsigned long long)kk * 16384ull;
      v4u g0 = { 1u,                                   // count=1, user desc
                 lds_base,                             // lds_addr
                 (unsigned)gaddr,                      // global_addr[31:0]
                 ((unsigned)(gaddr >> 32) & 0x01FFFFFFu) | 0x80000000u }; // [56:32] | type=2
      // data_size=8B(3), tensor_dim0=tile_dim0=2048, tensor_dim1=tile_dim1=1,
      // tensor_dim0_stride=2048
      v8u g1 = { 0x00030000u, 0x08000000u, 0x00010000u, 0x08000000u,
                 1u, 2048u, 0u, 0u };
      v4u gz = { 0u, 0u, 0u, 0u };
      asm volatile("tensor_load_to_lds %0, %1, %2, %3"
                   :: "s"(g0), "s"(g1), "s"(gz), "s"(gz)
                   : "memory");
    }

    // ---- cooperative bilinear sampling of tap kk into LDS (overlaps TDM) ----
    {
      float oy = offb[(2 * kk + 0) * HWD];
      float ox = offb[(2 * kk + 1) * HWD];
      float sy = (float)(ho + kk / 3) + oy;     // reference grid: ho + kh
      float sx = (float)(n_s + kk % 3) + ox;    // reference grid: wo + kw
      float y0f = floorf(sy), x0f = floorf(sx);
      float wy = sy - y0f, wx = sx - x0f;
      int iy0 = (int)y0f, ix0 = (int)x0f;
      int iy1 = iy0 + 1, ix1 = ix0 + 1;
      float vy0 = (iy0 >= 0 && iy0 < HD) ? 1.f : 0.f;
      float vy1 = (iy1 >= 0 && iy1 < HD) ? 1.f : 0.f;
      float vx0 = (ix0 >= 0 && ix0 < WD) ? 1.f : 0.f;
      float vx1 = (ix1 >= 0 && ix1 < WD) ? 1.f : 0.f;
      float w00 = (1.f - wy) * (1.f - wx) * vy0 * vx0;
      float w01 = (1.f - wy) * wx * vy0 * vx1;
      float w10 = wy * (1.f - wx) * vy1 * vx0;
      float w11 = wy * wx * vy1 * vx1;
      int cy0 = min(max(iy0, 0), HD - 1), cy1 = min(max(iy1, 0), HD - 1);
      int cx0 = min(max(ix0, 0), WD - 1), cx1 = min(max(ix1, 0), WD - 1);
      int i00 = cy0 * WD + cx0, i01 = cy0 * WD + cx1;
      int i10 = cy1 * WD + cx0, i11 = cy1 * WD + cx1;
      const float* xb = x + ((size_t)b * CC + ch_s * 32) * HWD;
#pragma unroll 2
      for (int cc = 0; cc < 32; cc += 2) {
        const float* xc0 = xb + cc * HWD;
        const float* xc1 = xc0 + HWD;
        float v0 = w00 * xc0[i00] + w01 * xc0[i01] + w10 * xc0[i10] + w11 * xc0[i11];
        float v1 = w00 * xc1[i00] + w01 * xc1[i01] + w10 * xc1[i10] + w11 * xc1[i11];
        unsigned h0 = f2bf(v0), h1 = f2bf(v1);
        unsigned l0 = f2bf(v0 - bf2f((unsigned short)h0));
        unsigned l1 = f2bf(v1 - bf2f((unsigned short)h1));
        int hb = (cc >> 3) & 1;                       // fragment lane-half
        int j  = (cc & 7) | ((cc & 16) >> 1);         // even -> b32-aligned
        *(unsigned*)&S[0][ch_s][hb][n_s][j] = h0 | (h1 << 16);
        *(unsigned*)&S[1][ch_s][hb][n_s][j] = l0 | (l1 << 16);
      }
    }

    if (wave == 0) __builtin_amdgcn_s_wait_tensorcnt(0);
    __syncthreads();   // S written + A_lds DMA complete

    // ---- WMMA: term-major for 4 independent accumulation chains ----
#pragma unroll
    for (int chunk = 0; chunk < 2; ++chunk) {
      union U { uint4 q[2]; v16bf v; };
      U bh, bl, ah[4], al;
      const unsigned short* sph = &S[0][chunk][lane >> 4][wave * 16 + (lane & 15)][0];
      const unsigned short* spl = &S[1][chunk][lane >> 4][wave * 16 + (lane & 15)][0];
      bh.q[0] = *(const uint4*)(sph);  bh.q[1] = *(const uint4*)(sph + 8);
      bl.q[0] = *(const uint4*)(spl);  bl.q[1] = *(const uint4*)(spl + 8);
#pragma unroll
      for (int mt = 0; mt < 4; ++mt) {
        const unsigned short* ap = &A_lds[0][chunk][mt][lane][0];
        ah[mt].q[0] = *(const uint4*)ap;  ah[mt].q[1] = *(const uint4*)(ap + 8);
      }
#pragma unroll
      for (int mt = 0; mt < 4; ++mt)
        acc[mt] = __builtin_amdgcn_wmma_f32_16x16x32_bf16(
            false, ah[mt].v, false, bh.v, (short)0, acc[mt], false, false);
#pragma unroll
      for (int mt = 0; mt < 4; ++mt)
        acc[mt] = __builtin_amdgcn_wmma_f32_16x16x32_bf16(
            false, ah[mt].v, false, bl.v, (short)0, acc[mt], false, false);
#pragma unroll
      for (int mt = 0; mt < 4; ++mt) {
        const unsigned short* ap = &A_lds[1][chunk][mt][lane][0];
        al.q[0] = *(const uint4*)ap;  al.q[1] = *(const uint4*)(ap + 8);
        acc[mt] = __builtin_amdgcn_wmma_f32_16x16x32_bf16(
            false, al.v, false, bh.v, (short)0, acc[mt], false, false);
      }
    }
  }

  // ---- epilogue: C/D layout -> (b, o, ho, wo); add bias ----
  int n  = wave * 16 + (lane & 15);
  int mo = (lane >> 4) * 8;
#pragma unroll
  for (int mt = 0; mt < 4; ++mt)
#pragma unroll
    for (int r = 0; r < 8; ++r) {
      int o = mt * 16 + mo + r;
      out[((size_t)b * OC + o) * HWD + ho * WD + n] = acc[mt][r] + bias[o];
    }
}

// ---------------------------------------------------------------------------
extern "C" void kernel_launch(void* const* d_in, const int* in_sizes, int n_in,
                              void* d_out, int out_size, void* d_ws, size_t ws_size,
                              hipStream_t stream) {
  const float* x      = (const float*)d_in[0];
  const float* weight = (const float*)d_in[1];
  const float* bias   = (const float*)d_in[2];
  const float* off_w  = (const float*)d_in[3];
  const float* off_b  = (const float*)d_in[4];
  float* out = (float*)d_out;

  // workspace layout: wfrag (147456 B) | off (9437184 B)
  unsigned short* wfrag = (unsigned short*)d_ws;
  float* off = (float*)((char*)d_ws + 147456);

  wprep_kernel<<<9, 256, 0, stream>>>(weight, wfrag);
  offconv_kernel<<<512, 256, 0, stream>>>(x, off_w, off_b, off);
  dconv_wmma_kernel<<<1024, 256, 0, stream>>>(x, bias, off, wfrag, out);
}
